// Generator_8847632630347
// MI455X (gfx1250) — matrix operand order repaired and checked by reading, not compiled, not run on hardware
//
#include <hip/hip_runtime.h>
#include <stdint.h>

// ---------------------------------------------------------------------------
// Problem constants (match reference)
// ---------------------------------------------------------------------------
#define N_NODES 50000
#define N_EDGES 800000
#define DIM     128     // D
#define HIDN    64      // H
#define HID2    128     // 2H (edge-rep K)
#define HID4    256     // 4H (edge-MLP N)
#define NGRAPH  128     // G
#define NPAD    (1 << 20)   // bitonic sort size (>= N_EDGES)

// Workspace layout (bytes)
#define WS_HF16   0ull                                   // N * 64 f16   = 6.4 MB
#define WS_W1F16  6400000ull                             // 128*256 f16  = 64 KB (swizzled)
#define WS_SEG    (WS_W1F16 + 65536ull)                  // E int        = 3.2 MB
#define WS_KEYS   (WS_SEG + 4ull * N_EDGES)              // NPAD u64     = 8 MB
#define WS_DEG    (WS_KEYS + 8ull * NPAD)                // 128 u32
#define WS_KSEL   (WS_DEG + 512ull)                      // 128 u32
#define WS_START  (WS_KSEL + 512ull)                     // 128 u32
#define WS_MINMAX (WS_START + 512ull)                    // 2 u32

typedef _Float16 v16h __attribute__((ext_vector_type(16)));
typedef float    v8f  __attribute__((ext_vector_type(8)));

__device__ __forceinline__ unsigned ordered_u32(float f) {
  unsigned b = __float_as_uint(f);
  return (b & 0x80000000u) ? ~b : (b | 0x80000000u);
}
__device__ __forceinline__ float ordered_dec(unsigned u) {
  unsigned b = (u & 0x80000000u) ? (u & 0x7FFFFFFFu) : ~u;
  return __uint_as_float(b);
}

// A-fragment scatter helper: for an 8-half chunk starting at K-index k0
// (k0 multiple of 8) of row m, returns the halfword offset inside a
// fragment-major A buffer laid out as [kk][lane][16 halfs].
// A 16x32 layout (contiguous, matches B): lane = hlf*16 + m holds K = kk*32 +
// {hlf*16..hlf*16+15} (elems 0-15).
__device__ __forceinline__ int a_frag_chunk_off(int m, int k0) {
  int kk   = k0 >> 5;
  int sub  = k0 & 31;
  int hlf  = sub >> 4;
  int grp  = (sub >> 3) & 1;
  int lane = hlf * 16 + m;
  return (kk * 32 + lane) * 16 + grp * 8;
}

// ---------------------------------------------------------------------------
// Kernel A: h = leaky_relu(x @ enc_w + enc_b), stored as f16 [N, 64]
// One wave -> one 16x64 output tile via v_wmma_f32_16x16x32_f16.
// Fragments are contiguous 32B per lane (ds_load_b128 pairs); the C tile is
// transposed through LDS so global writeback is 4x global_store_b128 per lane.
// ---------------------------------------------------------------------------
__global__ __launch_bounds__(256) void encoder_kernel(
    const float* __restrict__ x, const float* __restrict__ enc_w,
    const float* __restrict__ enc_b, _Float16* __restrict__ hf16)
{
  // enc_w swizzled fragment-major: [kk(4)][nt(4)][lane(32)][16 halfs] = 16 KB
  __shared__ _Float16 s_w[4 * 4 * 32 * 16];
  // per-wave A tile / output staging: 4 KB per wave
  __shared__ _Float16 s_a[8][4 * 32 * 16];
  __shared__ float    s_b[HIDN];

  const int t = threadIdx.x;
  // Swizzle enc_w (B layout: lanes 0-15 K=i, lanes 16-31 K=16+i, N=lane&15)
  for (int idx = t; idx < DIM * HIDN; idx += 256) {
    int k = idx >> 6, n = idx & 63;
    int kk = k >> 5, sub = k & 31;
    int hl = sub >> 4, i = sub & 15;
    int lane = hl * 16 + (n & 15), nt = n >> 4;
    s_w[((kk * 4 + nt) * 32 + lane) * 16 + i] = (_Float16)enc_w[idx];
  }
  if (t < HIDN) s_b[t] = enc_b[t];
  __syncthreads();

  const int wave = t >> 5, lane = t & 31;
  const int hlf = lane >> 4, m16 = lane & 15;
  const int row0 = (blockIdx.x * 8 + wave) * 16;
  _Float16* awz = s_a[wave];

  // Stage 16 rows of x as f16 directly into A-fragment layout.
  // Lane pair: m = lane>>1 (row), coff = (lane&1)*64 (column half).
  {
    int m    = lane >> 1;
    int coff = (lane & 1) * 64;
    int grow = row0 + m; if (grow >= N_NODES) grow = N_NODES - 1;
    const float* src = x + (size_t)grow * DIM + coff;
#pragma unroll
    for (int c8 = 0; c8 < 8; ++c8) {
      float4 lo = *(const float4*)(src + c8 * 8);
      float4 hi = *(const float4*)(src + c8 * 8 + 4);
      union { _Float16 h[8]; uint4 v; } u;
      u.h[0] = (_Float16)lo.x; u.h[1] = (_Float16)lo.y;
      u.h[2] = (_Float16)lo.z; u.h[3] = (_Float16)lo.w;
      u.h[4] = (_Float16)hi.x; u.h[5] = (_Float16)hi.y;
      u.h[6] = (_Float16)hi.z; u.h[7] = (_Float16)hi.w;
      *(uint4*)&awz[a_frag_chunk_off(m, coff + c8 * 8)] = u.v;
    }
  }
  __syncthreads();

  // Contiguous 32B fragment loads (2x ds_load_b128 each)
  v16h a[4];
#pragma unroll
  for (int kk = 0; kk < 4; ++kk)
    a[kk] = *(const v16h*)&awz[(kk * 32 + lane) * 16];

#pragma unroll
  for (int nt = 0; nt < 4; ++nt) {
    v8f c = {};
#pragma unroll
    for (int kk = 0; kk < 4; ++kk) {
      v16h b = *(const v16h*)&s_w[((kk * 4 + nt) * 32 + lane) * 16];
      c = __builtin_amdgcn_wmma_f32_16x16x32_f16(false, a[kk], false, b,
                                                 (short)0, c, false, false);
    }
    float bias = s_b[nt * 16 + m16];
    // Stage results row-major into the (consumed) A buffer: [16 rows][64 cols]
#pragma unroll
    for (int r = 0; r < 8; ++r) {          // C layout: M = r + 8*hlf, N = lane&15
      float v = c[r] + bias;
      v = fmaxf(v, 0.01f * v);             // leaky_relu
      awz[(r + 8 * hlf) * HIDN + nt * 16 + m16] = (_Float16)v;
    }
  }
  __syncthreads();

  // Coalesced writeback: lane pair covers one row (32 halfs = 64B per lane)
  {
    int m    = lane >> 1;
    int coff = (lane & 1) * 32;
    int grow = row0 + m;
    if (grow < N_NODES) {
      const uint4* src = (const uint4*)&awz[m * HIDN + coff];
      uint4*       dst = (uint4*)(hf16 + (size_t)grow * HIDN + coff);
#pragma unroll
      for (int i = 0; i < 4; ++i) dst[i] = src[i];
    }
  }
}

// ---------------------------------------------------------------------------
// w1 fp32 -> f16, swizzled fragment-major: [kk(4)][nt(16)][lane(32)][16 halfs]
// ---------------------------------------------------------------------------
__global__ void convert_w1_kernel(const float* __restrict__ w1,
                                  _Float16* __restrict__ w1s) {
  int idx = blockIdx.x * 256 + threadIdx.x;
  if (idx < HID2 * HID4) {
    int k = idx >> 8, n = idx & 255;       // w1 is [128][256] row-major
    int kk = k >> 5, sub = k & 31;
    int hl = sub >> 4, i = sub & 15;
    int lane = hl * 16 + (n & 15), nt = n >> 4;
    w1s[((kk * 16 + nt) * 32 + lane) * 16 + i] = (_Float16)w1[idx];
  }
}

// ---------------------------------------------------------------------------
// Kernel B (fused edge MLP): score = relu([h[row],h[col]] @ W1 + b1) @ w2 + b2
// One wave -> 16 edges, 64 WMMA per wave; hidden activations never leave
// accumulator VGPRs. All fragment LDS traffic is b128-wide.
// ---------------------------------------------------------------------------
__global__ __launch_bounds__(256) void edge_mlp_kernel(
    const _Float16* __restrict__ hf16, const _Float16* __restrict__ w1s,
    const int* __restrict__ edge_index, const float* __restrict__ att1_b,
    const float* __restrict__ att2_w, const float* __restrict__ att2_b,
    float* __restrict__ score)
{
  extern __shared__ char smem[];
  _Float16* s_w1 = (_Float16*)smem;                      // 64 KB (pre-swizzled)
  _Float16* s_a  = (_Float16*)(smem + 65536);            // 8 waves * 4 KB
  float*    s_b1 = (float*)(smem + 65536 + 32768);       // 256 f32
  float*    s_w2 = s_b1 + HID4;                          // 256 f32
  float*    s_sc = s_w2 + HID4;                          // 8 waves * 16 f32

  const int t = threadIdx.x;
  // Linear 16B-wide copy of the swizzled W1 into LDS (L2-resident source)
  {
    const uint4* src = (const uint4*)w1s;
    uint4*       dst = (uint4*)s_w1;
    for (int i = t; i < (HID2 * HID4) / 8; i += 256) dst[i] = src[i];
  }
  s_b1[t] = att1_b[t];
  s_w2[t] = att2_w[t];
  __syncthreads();

  const int wave = t >> 5, lane = t & 31;
  const int hlf = lane >> 4, m16 = lane & 15;
  const int e0 = (blockIdx.x * 8 + wave) * 16;
  _Float16* awz = s_a + wave * (4 * 32 * 16);

  // Gather the 16x128 edge-rep tile straight into A-fragment layout.
  // Lane pair: m = lane>>1 (edge), endp = lane&1 (row/col endpoint).
  {
    int m    = lane >> 1;
    int endp = lane & 1;
    int e    = e0 + m;
    __builtin_prefetch(&edge_index[(size_t)endp * N_EDGES + e + 2048], 0, 1);
    int node = edge_index[(size_t)endp * N_EDGES + e];
    const uint4* src = (const uint4*)(hf16 + (size_t)node * HIDN);
#pragma unroll
    for (int c8 = 0; c8 < 8; ++c8) {
      uint4 v = src[c8];                               // 8 halfs of h[node]
      *(uint4*)&awz[a_frag_chunk_off(m, endp * 64 + c8 * 8)] = v;
    }
  }
  __syncthreads();

  // A fragments over K = 128: 4 contiguous 32B loads
  v16h a[4];
#pragma unroll
  for (int kk = 0; kk < 4; ++kk)
    a[kk] = *(const v16h*)&awz[(kk * 32 + lane) * 16];

  float sc[8];
#pragma unroll
  for (int r = 0; r < 8; ++r) sc[r] = 0.0f;

  for (int nt = 0; nt < 16; ++nt) {        // 16 tiles of N=16 -> 4H = 256
    v8f c = {};
#pragma unroll
    for (int kk = 0; kk < 4; ++kk) {
      v16h b = *(const v16h*)&s_w1[((kk * 16 + nt) * 32 + lane) * 16];
      c = __builtin_amdgcn_wmma_f32_16x16x32_f16(false, a[kk], false, b,
                                                 (short)0, c, false, false);
    }
    // Fused bias + relu + dot with att2_w, all in-register
    float bias = s_b1[nt * 16 + m16];
    float w2v  = s_w2[nt * 16 + m16];
#pragma unroll
    for (int r = 0; r < 8; ++r)
      sc[r] += fmaxf(c[r] + bias, 0.0f) * w2v;
  }

  // Reduce across the 16 lanes sharing each row M (xor 1,2,4,8 stays in-half)
#pragma unroll
  for (int r = 0; r < 8; ++r) {
    float v = sc[r];
    v += __shfl_xor(v, 1, 32);
    v += __shfl_xor(v, 2, 32);
    v += __shfl_xor(v, 4, 32);
    v += __shfl_xor(v, 8, 32);
    sc[r] = v;
  }
  // Stage per-wave scores in LDS, then emit one coalesced b32 store per lane
  if (m16 == 0) {
    float b2 = att2_b[0];
#pragma unroll
    for (int r = 0; r < 8; ++r)
      s_sc[wave * 16 + hlf * 8 + r] = sc[r] + b2;  // lane0 -> M 0..7, lane16 -> 8..15
  }
  __syncthreads();
  if (lane < 16) score[e0 + lane] = s_sc[wave * 16 + lane];
}

// ---------------------------------------------------------------------------
// Reduction / sort / mask pipeline
// ---------------------------------------------------------------------------
__global__ void init_stats_kernel(unsigned* __restrict__ deg,
                                  unsigned* __restrict__ minmax) {
  int t = threadIdx.x;
  if (t < NGRAPH) deg[t] = 0u;
  if (t == 0) { minmax[0] = 0xFFFFFFFFu; minmax[1] = 0u; }
}

__global__ __launch_bounds__(256) void edge_stats_kernel(
    const float* __restrict__ score, const int* __restrict__ edge_index,
    const int* __restrict__ batch, int* __restrict__ seg,
    unsigned* __restrict__ deg, unsigned* __restrict__ minmax)
{
  __shared__ unsigned smin[256], smax[256];
  int e = blockIdx.x * 256 + threadIdx.x;
  unsigned u = ordered_u32(score[e]);
  int g = batch[edge_index[e]];            // seg = batch[row[e]]
  seg[e] = g;
  atomicAdd(&deg[g], 1u);
  smin[threadIdx.x] = u; smax[threadIdx.x] = u;
  __syncthreads();
  for (int s = 128; s > 0; s >>= 1) {
    if (threadIdx.x < s) {
      smin[threadIdx.x] = min(smin[threadIdx.x], smin[threadIdx.x + s]);
      smax[threadIdx.x] = max(smax[threadIdx.x], smax[threadIdx.x + s]);
    }
    __syncthreads();
  }
  if (threadIdx.x == 0) {
    atomicMin(&minmax[0], smin[0]);
    atomicMax(&minmax[1], smax[0]);
  }
}

__global__ void scan_kernel(const unsigned* __restrict__ deg,
                            unsigned* __restrict__ ksel,
                            unsigned* __restrict__ start)
{
  __shared__ unsigned s[NGRAPH];
  int t = threadIdx.x;                     // 128 threads
  unsigned d = deg[t];
  ksel[t] = (d + 3u) >> 2;                 // ceil(0.25 * deg)
  s[t] = d;
  __syncthreads();
  for (int off = 1; off < NGRAPH; off <<= 1) {
    unsigned v = (t >= off) ? s[t - off] : 0u;
    __syncthreads();
    s[t] += v;
    __syncthreads();
  }
  start[t] = s[t] - d;                     // exclusive cumsum
}

__global__ __launch_bounds__(256) void make_keys_kernel(
    const float* __restrict__ score, const int* __restrict__ seg,
    const unsigned* __restrict__ minmax,
    unsigned long long* __restrict__ keys)
{
  int i = blockIdx.x * 256 + threadIdx.x;
  unsigned long long k;
  if (i < N_EDGES) {
    float fmin = ordered_dec(minmax[0]);
    float fmax = ordered_dec(minmax[1]);
    float norm = (score[i] - fmin) / (fmax - fmin + 1e-12f) - (float)seg[i];
    unsigned ku = ~ordered_u32(norm);      // inverted -> ascending sort == descending norm
    k = ((unsigned long long)ku << 32) | (unsigned long long)(unsigned)i;
  } else {
    k = 0xFFFFFFFFFFFFFFFFull;             // padding sorts to the end
  }
  keys[i] = k;
}

__global__ __launch_bounds__(256) void bitonic_kernel(
    unsigned long long* __restrict__ keys, int j, int k)
{
  int i = blockIdx.x * 256 + threadIdx.x;
  int l = i ^ j;
  if (l > i) {
    unsigned long long a = keys[i], b = keys[l];
    bool up = ((i & k) == 0);
    if ((a > b) == up) { keys[i] = b; keys[l] = a; }
  }
}

__global__ __launch_bounds__(256) void finalize_kernel(
    const unsigned long long* __restrict__ keys, const int* __restrict__ seg,
    const unsigned* __restrict__ ksel, const unsigned* __restrict__ start,
    float* __restrict__ out_perm, float* __restrict__ out_mask)
{
  int i = blockIdx.x * 256 + threadIdx.x;  // i < E: real entries sort first
  unsigned e = (unsigned)(keys[i] & 0xFFFFFFFFull);
  int g = seg[e];
  unsigned within = (unsigned)i - start[g];
  out_perm[i] = (float)e;
  out_mask[i] = (within < ksel[g]) ? 1.0f : 0.0f;
}

// ---------------------------------------------------------------------------
// Host launcher
// ---------------------------------------------------------------------------
extern "C" void kernel_launch(void* const* d_in, const int* in_sizes, int n_in,
                              void* d_out, int out_size, void* d_ws, size_t ws_size,
                              hipStream_t stream) {
  const float* x      = (const float*)d_in[0];
  const int*   ei     = (const int*)d_in[1];
  const int*   batch  = (const int*)d_in[2];
  const float* enc_w  = (const float*)d_in[3];
  const float* enc_b  = (const float*)d_in[4];
  const float* att1_w = (const float*)d_in[5];
  const float* att1_b = (const float*)d_in[6];
  const float* att2_w = (const float*)d_in[7];
  const float* att2_b = (const float*)d_in[8];

  char* ws = (char*)d_ws;
  _Float16* hf16   = (_Float16*)(ws + WS_HF16);
  _Float16* w1s    = (_Float16*)(ws + WS_W1F16);
  int*      seg    = (int*)(ws + WS_SEG);
  unsigned long long* keys = (unsigned long long*)(ws + WS_KEYS);
  unsigned* deg    = (unsigned*)(ws + WS_DEG);
  unsigned* ksel   = (unsigned*)(ws + WS_KSEL);
  unsigned* start  = (unsigned*)(ws + WS_START);
  unsigned* minmax = (unsigned*)(ws + WS_MINMAX);

  float* out_score = (float*)d_out;
  float* out_perm  = out_score + N_EDGES;
  float* out_mask  = out_perm + N_EDGES;

  hipLaunchKernelGGL(init_stats_kernel, dim3(1), dim3(128), 0, stream, deg, minmax);
  hipLaunchKernelGGL(convert_w1_kernel, dim3((HID2 * HID4 + 255) / 256), dim3(256),
                     0, stream, att1_w, w1s);
  hipLaunchKernelGGL(encoder_kernel, dim3((N_NODES + 127) / 128), dim3(256),
                     0, stream, x, enc_w, enc_b, hf16);

  size_t smemB = 65536 + 32768 + 2 * HID4 * sizeof(float) + 8 * 16 * sizeof(float);
  hipLaunchKernelGGL(edge_mlp_kernel, dim3(N_EDGES / 128), dim3(256), smemB, stream,
                     hf16, w1s, ei, att1_b, att2_w, att2_b, out_score);

  hipLaunchKernelGGL(edge_stats_kernel, dim3(N_EDGES / 256), dim3(256), 0, stream,
                     out_score, ei, batch, seg, deg, minmax);
  hipLaunchKernelGGL(scan_kernel, dim3(1), dim3(NGRAPH), 0, stream, deg, ksel, start);
  hipLaunchKernelGGL(make_keys_kernel, dim3(NPAD / 256), dim3(256), 0, stream,
                     out_score, seg, minmax, keys);

  for (int k = 2; k <= NPAD; k <<= 1)
    for (int j = k >> 1; j > 0; j >>= 1)
      hipLaunchKernelGGL(bitonic_kernel, dim3(NPAD / 256), dim3(256), 0, stream,
                         keys, j, k);

  hipLaunchKernelGGL(finalize_kernel, dim3(N_EDGES / 256), dim3(256), 0, stream,
                     keys, seg, ksel, start, out_perm, out_mask);
}
